// MultiHeadCausalSelfAttention_44813688766609
// MI455X (gfx1250) — matrix operand order repaired and checked by reading, not compiled, not run on hardware
//
#include <hip/hip_runtime.h>
#include <hip/hip_bf16.h>
#include <stdint.h>

typedef __bf16 bf16;
typedef __attribute__((ext_vector_type(16))) __bf16 v16bf;
typedef __attribute__((ext_vector_type(8)))  float  v8f;

#define EMBED 1024
#define SEQ   2048
#define HEADS 16
#define DHEAD 64
#define BATCH 2
#define TOKENS (BATCH*SEQ)

union FragAB { v16bf v; uint4 q[2]; };

// A fragment (16x32 bf16, MxK). ISA layout: lanes 0-15 -> rows M=0..15, K {0-7,16-23};
// lanes 16-31 -> same rows, K {8-15,24-31}. LDS tile is [row][k], 'stride' elements/row.
__device__ __forceinline__ v16bf ldsA(const bf16* base, int stride, int r0, int lane) {
  int row = r0 + (lane & 15);
  int kh  = (lane & 16) ? 8 : 0;
  const bf16* p = base + row * stride + kh;
  FragAB u;
  u.q[0] = *(const uint4*)(p);        // K kh..kh+7
  u.q[1] = *(const uint4*)(p + 16);   // K kh+16..kh+23
  return u.v;
}
// B fragment (32x16 bf16, KxN). Lanes 0-15 hold K {0-7,16-23}, lanes 16-31 hold K {8-15,24-31},
// lane column N = lane&15. LDS tile stored as [n][k], k contiguous.
__device__ __forceinline__ v16bf ldsB(const bf16* base, int stride, int n0, int lane) {
  int n  = n0 + (lane & 15);
  int kh = (lane & 16) ? 8 : 0;
  const bf16* p = base + n * stride + kh;
  FragAB u;
  u.q[0] = *(const uint4*)(p);
  u.q[1] = *(const uint4*)(p + 16);
  return u.v;
}

__device__ __forceinline__ v8f wmma_bf16(v16bf a, v16bf b, v8f c) {
  return __builtin_amdgcn_wmma_f32_16x16x32_bf16(false, a, false, b, (short)0, c, false, false);
}

// Pack 8 consecutive fp32 -> 8 bf16 (one 16B LDS store)
__device__ __forceinline__ uint4 pack8(const float* g) {
  float4 f0 = *(const float4*)(g);
  float4 f1 = *(const float4*)(g + 4);
  union { bf16 h[8]; uint4 q; } pk;
  pk.h[0]=(bf16)f0.x; pk.h[1]=(bf16)f0.y; pk.h[2]=(bf16)f0.z; pk.h[3]=(bf16)f0.w;
  pk.h[4]=(bf16)f1.x; pk.h[5]=(bf16)f1.y; pk.h[6]=(bf16)f1.z; pk.h[7]=(bf16)f1.w;
  return pk.q;
}

// CDNA5 async copy: 16 bytes global -> LDS through the async engine (ASYNCcnt).
// vdst carries the wave-relative LDS byte offset = low 32 bits of the generic pointer.
__device__ __forceinline__ void async_copy16(const bf16* g, bf16* l) {
  asm volatile("global_load_async_to_lds_b128 %0, %1, off"
               :: "v"((uint32_t)(uintptr_t)l), "v"(g) : "memory");
}
__device__ __forceinline__ void wait_async() {
  asm volatile("s_wait_asynccnt 0" ::: "memory");
}

// ---------------------------------------------------------------------------
// Kernel 1: Y = (X @ W^T + bias) * scale -> bf16 in [b][h][s][d] layout.
// Block 256 threads (8 waves). Tile 128 (tokens) x 128 (embed), K-chunk 64.
// Each wave owns a 32x64 sub-tile (2x4 WMMA frags): 16 WMMAs per staging round.
// ---------------------------------------------------------------------------
__global__ __launch_bounds__(256)
void qkv_gemm_kernel(const float* __restrict__ X, const float* __restrict__ W,
                     const float* __restrict__ bias, bf16* __restrict__ out, float scale) {
  __shared__ bf16 sX[128 * 72];
  __shared__ bf16 sW[128 * 72];
  const int tid = threadIdx.x, lane = tid & 31, wave = tid >> 5;
  const int m0 = blockIdx.x * 128;   // token rows
  const int n0 = blockIdx.y * 128;   // embed cols
  const int rs = (wave >> 1) * 32;   // wave row sub-tile
  const int cs = (wave & 1) * 64;    // wave col sub-tile
  v8f acc[2][4] = {};

  const int xr = tid >> 1, xc = (tid & 1) * 32;  // 32 elements per thread per tile

  for (int kk = 0; kk < EMBED; kk += 64) {
    { // stage X tile 128x64 fp32 -> bf16
      const float* g = X + (size_t)(m0 + xr) * EMBED + kk + xc;
      bf16* l = &sX[xr * 72 + xc];
      *(uint4*)(l)      = pack8(g);
      *(uint4*)(l + 8)  = pack8(g + 8);
      *(uint4*)(l + 16) = pack8(g + 16);
      *(uint4*)(l + 24) = pack8(g + 24);
    }
    { // stage W tile 128x64 fp32 -> bf16 (rows = output cols m, k contiguous)
      const float* g = W + (size_t)(n0 + xr) * EMBED + kk + xc;
      bf16* l = &sW[xr * 72 + xc];
      *(uint4*)(l)      = pack8(g);
      *(uint4*)(l + 8)  = pack8(g + 8);
      *(uint4*)(l + 16) = pack8(g + 16);
      *(uint4*)(l + 24) = pack8(g + 24);
    }
    __syncthreads();
#pragma unroll
    for (int c = 0; c < 2; ++c) {
      v16bf a0 = ldsA(sX + c * 32, 72, rs,      lane);
      v16bf a1 = ldsA(sX + c * 32, 72, rs + 16, lane);
#pragma unroll
      for (int j = 0; j < 4; ++j) {
        v16bf b = ldsB(sW + c * 32, 72, cs + j * 16, lane);
        acc[0][j] = wmma_bf16(a0, b, acc[0][j]);
        acc[1][j] = wmma_bf16(a1, b, acc[1][j]);
      }
    }
    __syncthreads();
  }

  const int cgrp = lane >> 4, cn = lane & 15;
#pragma unroll
  for (int i = 0; i < 2; ++i) {
#pragma unroll
    for (int j = 0; j < 4; ++j) {
      int m = n0 + cs + j * 16 + cn;                 // embed col
      float bb = bias[m];
      int h = m >> 6, d = m & 63;
#pragma unroll
      for (int v = 0; v < 8; ++v) {
        int n = m0 + rs + i * 16 + v + 8 * cgrp;     // token
        int b = n >> 11, s = n & (SEQ - 1);
        float val = (acc[i][j][v] + bb) * scale;
        out[(((size_t)(b * HEADS + h)) * SEQ + s) * DHEAD + d] = (bf16)val;
      }
    }
  }
}

// ---------------------------------------------------------------------------
// Kernel 2: flash attention. Block = 128 threads (4 waves), 64 query rows per
// block, one (b,h) per blockIdx.y, key tiles of 64, causal. K tile staged via
// async-to-LDS; V staged transposed; Q fragments register-resident.
// ---------------------------------------------------------------------------
__global__ __launch_bounds__(128)
void flash_kernel(const bf16* __restrict__ Q, const bf16* __restrict__ K,
                  const bf16* __restrict__ V, bf16* __restrict__ O) {
  __shared__ bf16 sK[64 * 72];        // [key][d]
  __shared__ bf16 sV[64 * 72];        // transposed: [d][key]
  __shared__ bf16 sP[4 * 16 * 72];    // per-wave P strip [row][key]
  const int tid = threadIdx.x, lane = tid & 31, wave = tid >> 5;
  const int qt = blockIdx.x;
  const int bh = blockIdx.y;
  const int bId = bh >> 4, hId = bh & 15;
  const size_t baseBH = (size_t)bh * SEQ * DHEAD;
  const int qr = qt * 64 + wave * 16;
  const int cgrp = lane >> 4, cn = lane & 15;

  v16bf aq[2];
#pragma unroll
  for (int c = 0; c < 2; ++c) {
    const bf16* p = Q + baseBH + (size_t)(qr + (lane & 15)) * DHEAD + c * 32 + ((lane & 16) ? 8 : 0);
    FragAB u;
    u.q[0] = *(const uint4*)(p);
    u.q[1] = *(const uint4*)(p + 16);
    aq[c] = u.v;
  }

  float mrow[8], lrow[8];
  v8f oacc[4] = {};
#pragma unroll
  for (int v = 0; v < 8; ++v) { mrow[v] = -3.0e38f; lrow[v] = 0.0f; }

  bf16* sPw = sP + wave * 16 * 72;

  for (int kt = 0; kt <= qt; ++kt) {
    const int key = tid >> 1, dc = (tid & 1) * 32;
    { // stage K tile [key][d] via async engine
      const bf16* g = K + baseBH + (size_t)(kt * 64 + key) * DHEAD + dc;
      bf16* l = &sK[key * 72 + dc];
      async_copy16(g,      l);
      async_copy16(g + 8,  l + 8);
      async_copy16(g + 16, l + 16);
      async_copy16(g + 24, l + 24);
    }
    { // stage V transposed [d][key] (conversion of layout -> manual path)
      const uint4* g = (const uint4*)(V + baseBH + (size_t)(kt * 64 + key) * DHEAD + dc);
      union { uint4 q[4]; bf16 h[32]; } u;
      u.q[0] = g[0]; u.q[1] = g[1]; u.q[2] = g[2]; u.q[3] = g[3];
#pragma unroll
      for (int i = 0; i < 32; ++i) sV[(dc + i) * 72 + key] = u.h[i];
    }
    wait_async();
    __syncthreads();

    // scores = Q @ K^T (Q pre-scaled by 1/sqrt(dh))
    v8f sc[4] = {};
#pragma unroll
    for (int c = 0; c < 2; ++c) {
#pragma unroll
      for (int j = 0; j < 4; ++j) {
        v16bf bk = ldsB(sK + c * 32, 72, j * 16, lane);
        sc[j] = wmma_bf16(aq[c], bk, sc[j]);
      }
    }
    if (kt == qt) {   // causal mask on diagonal tile
#pragma unroll
      for (int j = 0; j < 4; ++j) {
        int colg = kt * 64 + j * 16 + cn;
#pragma unroll
        for (int v = 0; v < 8; ++v) {
          int rowg = qr + v + 8 * cgrp;
          if (colg > rowg) sc[j][v] = -3.0e38f;
        }
      }
    }

    // online softmax (rows live in 16-lane groups of C layout)
#pragma unroll
    for (int v = 0; v < 8; ++v) {
      float t = fmaxf(fmaxf(sc[0][v], sc[1][v]), fmaxf(sc[2][v], sc[3][v]));
      t = fmaxf(t, __shfl_xor(t, 1, 16));
      t = fmaxf(t, __shfl_xor(t, 2, 16));
      t = fmaxf(t, __shfl_xor(t, 4, 16));
      t = fmaxf(t, __shfl_xor(t, 8, 16));
      float mnew  = fmaxf(mrow[v], t);
      float alpha = __expf(mrow[v] - mnew);
      float rsum  = 0.0f;
#pragma unroll
      for (int j = 0; j < 4; ++j) {
        float p = __expf(sc[j][v] - mnew);
        sc[j][v] = p;
        rsum += p;
      }
      rsum += __shfl_xor(rsum, 1, 16);
      rsum += __shfl_xor(rsum, 2, 16);
      rsum += __shfl_xor(rsum, 4, 16);
      rsum += __shfl_xor(rsum, 8, 16);
      lrow[v] = lrow[v] * alpha + rsum;
      mrow[v] = mnew;
#pragma unroll
      for (int jd = 0; jd < 4; ++jd) oacc[jd][v] = oacc[jd][v] * alpha;
    }

    // C-layout -> A-layout via per-wave LDS strip
#pragma unroll
    for (int j = 0; j < 4; ++j)
#pragma unroll
      for (int v = 0; v < 8; ++v)
        sPw[(v + 8 * cgrp) * 72 + j * 16 + cn] = (bf16)sc[j][v];
    asm volatile("s_wait_dscnt 0" ::: "memory");

    // O += P @ V
#pragma unroll
    for (int c = 0; c < 2; ++c) {
      v16bf ap = ldsA(sPw + c * 32, 72, 0, lane);
#pragma unroll
      for (int jd = 0; jd < 4; ++jd) {
        v16bf bvf = ldsB(sV + c * 32, 72, jd * 16, lane);
        oacc[jd] = wmma_bf16(ap, bvf, oacc[jd]);
      }
    }
    __syncthreads();
  }

  // normalize; write attn output bf16 [b][s][h*64+d] (TOKENS x EMBED row-major)
#pragma unroll
  for (int v = 0; v < 8; ++v) {
    float inv = 1.0f / lrow[v];
    int s = qr + v + 8 * cgrp;
    size_t n = (size_t)bId * SEQ + s;
#pragma unroll
    for (int jd = 0; jd < 4; ++jd) {
      int m = hId * 64 + jd * 16 + cn;
      O[n * EMBED + m] = (bf16)(oacc[jd][v] * inv);
    }
  }
}

// ---------------------------------------------------------------------------
// Kernel 3: out = A(bf16) @ Wo^T + bo -> fp32. A tile staged via async-to-LDS.
// ---------------------------------------------------------------------------
__global__ __launch_bounds__(256)
void out_gemm_kernel(const bf16* __restrict__ A, const float* __restrict__ W,
                     const float* __restrict__ bias, float* __restrict__ out) {
  __shared__ bf16 sX[128 * 72];
  __shared__ bf16 sW[128 * 72];
  const int tid = threadIdx.x, lane = tid & 31, wave = tid >> 5;
  const int m0 = blockIdx.x * 128;
  const int n0 = blockIdx.y * 128;
  const int rs = (wave >> 1) * 32;
  const int cs = (wave & 1) * 64;
  v8f acc[2][4] = {};

  const int xr = tid >> 1, xc = (tid & 1) * 32;

  for (int kk = 0; kk < EMBED; kk += 64) {
    { // stage A tile (already bf16): async global->LDS
      const bf16* g = A + (size_t)(m0 + xr) * EMBED + kk + xc;
      bf16* l = &sX[xr * 72 + xc];
      async_copy16(g,      l);
      async_copy16(g + 8,  l + 8);
      async_copy16(g + 16, l + 16);
      async_copy16(g + 24, l + 24);
    }
    { // stage W tile fp32 -> bf16
      const float* g = W + (size_t)(n0 + xr) * EMBED + kk + xc;
      bf16* l = &sW[xr * 72 + xc];
      *(uint4*)(l)      = pack8(g);
      *(uint4*)(l + 8)  = pack8(g + 8);
      *(uint4*)(l + 16) = pack8(g + 16);
      *(uint4*)(l + 24) = pack8(g + 24);
    }
    wait_async();
    __syncthreads();
#pragma unroll
    for (int c = 0; c < 2; ++c) {
      v16bf a0 = ldsA(sX + c * 32, 72, rs,      lane);
      v16bf a1 = ldsA(sX + c * 32, 72, rs + 16, lane);
#pragma unroll
      for (int j = 0; j < 4; ++j) {
        v16bf b = ldsB(sW + c * 32, 72, cs + j * 16, lane);
        acc[0][j] = wmma_bf16(a0, b, acc[0][j]);
        acc[1][j] = wmma_bf16(a1, b, acc[1][j]);
      }
    }
    __syncthreads();
  }

  const int cgrp = lane >> 4, cn = lane & 15;
#pragma unroll
  for (int i = 0; i < 2; ++i) {
#pragma unroll
    for (int j = 0; j < 4; ++j) {
      int m = n0 + cs + j * 16 + cn;
      float bb = bias[m];
#pragma unroll
      for (int v = 0; v < 8; ++v) {
        int n = m0 + rs + i * 16 + v + 8 * cgrp;
        out[(size_t)n * EMBED + m] = acc[i][j][v] + bb;
      }
    }
  }
}

extern "C" void kernel_launch(void* const* d_in, const int* in_sizes, int n_in,
                              void* d_out, int out_size, void* d_ws, size_t ws_size,
                              hipStream_t stream) {
  (void)in_sizes; (void)n_in; (void)out_size; (void)ws_size;
  const float* x  = (const float*)d_in[0];
  // d_in[1] = mask (unused: causal mask applied analytically)
  const float* Wq = (const float*)d_in[2];
  const float* bq = (const float*)d_in[3];
  const float* Wk = (const float*)d_in[4];
  const float* bk = (const float*)d_in[5];
  const float* Wv = (const float*)d_in[6];
  const float* bv = (const float*)d_in[7];
  const float* Wo = (const float*)d_in[8];
  const float* bo = (const float*)d_in[9];

  bf16* ws = (bf16*)d_ws;
  const size_t per = (size_t)BATCH * HEADS * SEQ * DHEAD;  // 4,194,304 elements
  bf16* Qb = ws;
  bf16* Kb = ws + per;
  bf16* Vb = ws + 2 * per;
  bf16* Ab = ws + 3 * per;

  dim3 gg(TOKENS / 128, EMBED / 128);
  qkv_gemm_kernel<<<gg, 256, 0, stream>>>(x, Wq, bq, Qb, 0.125f);  // 1/sqrt(64) folded into Q
  qkv_gemm_kernel<<<gg, 256, 0, stream>>>(x, Wk, bk, Kb, 1.0f);
  qkv_gemm_kernel<<<gg, 256, 0, stream>>>(x, Wv, bv, Vb, 1.0f);
  flash_kernel<<<dim3(SEQ / 64, BATCH * HEADS), 128, 0, stream>>>(Qb, Kb, Vb, Ab);
  out_gemm_kernel<<<gg, 256, 0, stream>>>(Ab, Wo, bo, (float*)d_out);
}
